// CopyHead_90245852824125
// MI455X (gfx1250) — matrix operand order repaired and checked by reading, not compiled, not run on hardware
//
#include <hip/hip_runtime.h>

typedef _Float16 v8h  __attribute__((ext_vector_type(8)));
typedef _Float16 v16h __attribute__((ext_vector_type(16)));
typedef float    v8f  __attribute__((ext_vector_type(8)));

#define B_    8
#define T_    1024
#define K_    8
#define L_    1024
#define H_    256
#define E_    128
#define F_    64
#define V_    23
#define FEAT  448            // H + E + F (14 K-steps of 32)
#define TPB_T 8              // timesteps per workgroup
#define MROWS 64             // TPB_T * K_ (4 M-tiles of 16)
#define ASA   456            // LDS f16 row stride (mult. of 8 -> 16B aligned chunks,
                             // 228-dword rows spread across the 64 banks)
#define WTS   448            // W1T f16 row stride (mult. of 8)

// ---------- prep: W1 (448x256 f32, row-major K x N) -> W1T (256x448 f16, N x K) ----------
__global__ __launch_bounds__(256)
void prep_w1t(const float* __restrict__ W1, _Float16* __restrict__ W1T)
{
    const int idx = blockIdx.x * 256 + threadIdx.x;   // pair index, 256*224 total
    const int n   = idx / (FEAT / 2);
    const int kp  = (idx - n * (FEAT / 2)) * 2;
    if (n < H_) {
        union { _Float16 h[2]; unsigned u; } p;
        p.h[0] = (_Float16)W1[(kp    ) * H_ + n];
        p.h[1] = (_Float16)W1[(kp + 1) * H_ + n];
        *(unsigned*)&W1T[n * WTS + kp] = p.u;         // coalesced 4B stores along K
    }
}

// ---------------------------- fused copy-head kernel ----------------------------
__global__ __launch_bounds__(256)
void copyhead_fused(const float* __restrict__ hs,      // (B,T,H)
                    const float* __restrict__ ex,      // (B,K,L,E)
                    const float* __restrict__ colf,    // (B,L,F)
                    const int*   __restrict__ ctp,     // (B,T)
                    const int*   __restrict__ aa,      // (B,K,L)
                    const _Float16* __restrict__ W1T,  // (H, FEAT) f16, transposed
                    const float* __restrict__ b1p,     // (H)
                    const float* __restrict__ W2,      // (H,1)
                    const float* __restrict__ b2p,     // (1)
                    float* __restrict__ p_copy,        // (B,T,V)
                    float* __restrict__ lam)           // (B,T,K)
{
    __shared__ __align__(16) _Float16 Ah[MROWS * ASA];
    __shared__ float scoreS[MROWS];
    __shared__ float wbuf[MROWS];
    __shared__ int   aaS[MROWS];

    const int tid = threadIdx.x;
    const int blk = blockIdx.x;
    const int b   = blk / (T_ / TPB_T);
    const int t0  = (blk % (T_ / TPB_T)) * TPB_T;

    // ---------------- Phase 1: gather features -> f16 in LDS ----------------
    if (tid < MROWS) scoreS[tid] = b2p[0];            // bias b2 seeds score accum

    for (int p = tid; p < MROWS * (FEAT / 2); p += 256) {
        const int row = p / (FEAT / 2);
        const int cp  = (p - row * (FEAT / 2)) * 2;   // even column, never straddles segments
        const int tl  = row >> 3;
        const int k   = row & 7;
        const int t   = t0 + tl;
        const int cti = ctp[b * T_ + t];
        const int c   = cti < 0 ? 0 : cti;
        const float* src;
        if (cp < H_)            src = &hs[(b * T_ + t) * H_ + cp];
        else if (cp < H_ + E_)  src = &ex[(((b * K_ + k) * L_ + c) * E_) + (cp - H_)];
        else                    src = &colf[(b * L_ + c) * F_ + (cp - (H_ + E_))];
        union { _Float16 h[2]; unsigned u; } pk;
        pk.h[0] = (_Float16)src[0];
        pk.h[1] = (_Float16)src[1];
        *(unsigned*)&Ah[row * ASA + cp] = pk.u;
    }
    __syncthreads();

    // ---------------- Phase 2: WMMA f16 GEMM (feats @ W1 + b1) ----------------
    const int wv   = tid >> 5;                        // wave 0..7
    const int lane = tid & 31;
    const int half = lane >> 4;                       // CDNA5 16x16 lane group
    const int ln   = lane & 15;
    const int n0   = wv * 32;                         // wave's 32-column slab
    const int c0   = n0 + ln;
    const int c1   = n0 + 16 + ln;

    v8f acc[4][2];
    {
        const float bias0 = b1p[c0];
        const float bias1 = b1p[c1];
        #pragma unroll
        for (int m = 0; m < 4; ++m)
            #pragma unroll
            for (int i = 0; i < 8; ++i) { acc[m][0][i] = bias0; acc[m][1][i] = bias1; }
    }

    #pragma unroll 2
    for (int s = 0; s < FEAT / 32; ++s) {
        // A fragments (16x32 f16): per lane, K = 32s+8*half+{0..7} and +16 (two 16B chunks)
        const int ka = 32 * s + 8 * half;
        v16h a[4];
        #pragma unroll
        for (int m = 0; m < 4; ++m) {
            const v8h lo = *(const v8h*)&Ah[(m * 16 + ln) * ASA + ka];
            const v8h hi = *(const v8h*)&Ah[(m * 16 + ln) * ASA + ka + 16];
            a[m] = __builtin_shufflevector(lo, hi, 0, 1, 2, 3, 4, 5, 6, 7,
                                                   8, 9, 10, 11, 12, 13, 14, 15);
        }
        // B fragments (32x16 f16): per lane, col = c0/c1, K = 32s+8*half+{0..7} and +16 (two 16B chunks)
        const int kb = 32 * s + 8 * half;
        v16h bb[2];
        #pragma unroll
        for (int ns = 0; ns < 2; ++ns) {
            const _Float16* q = &W1T[(n0 + 16 * ns + ln) * WTS + kb];
            const v8h lo = *(const v8h*)q;
            const v8h hi = *(const v8h*)(q + 16);
            bb[ns] = __builtin_shufflevector(lo, hi, 0, 1, 2, 3, 4, 5, 6, 7,
                                                     8, 9, 10, 11, 12, 13, 14, 15);
        }
        #pragma unroll
        for (int m = 0; m < 4; ++m) {
            acc[m][0] = __builtin_amdgcn_wmma_f32_16x16x32_f16(false, a[m], false, bb[0],
                                                               (short)0, acc[m][0], false, false);
            acc[m][1] = __builtin_amdgcn_wmma_f32_16x16x32_f16(false, a[m], false, bb[1],
                                                               (short)0, acc[m][1], false, false);
        }
    }

    // -------- ReLU + h1 @ W2 (f32): butterfly row-sums + LDS atomic combine --------
    {
        const float w2a = W2[c0];
        const float w2b = W2[c1];
        #pragma unroll
        for (int m = 0; m < 4; ++m) {
            #pragma unroll
            for (int i = 0; i < 8; ++i) {
                float v = fmaxf(acc[m][0][i], 0.0f) * w2a +
                          fmaxf(acc[m][1][i], 0.0f) * w2b;
                v += __shfl_xor(v, 8, 32);
                v += __shfl_xor(v, 4, 32);
                v += __shfl_xor(v, 2, 32);
                v += __shfl_xor(v, 1, 32);
                if (ln == 0)
                    atomicAdd(&scoreS[m * 16 + i + 8 * half], v);   // ds_add_f32
            }
        }
    }
    __syncthreads();

    // ---------------- Phase 3: softmax over K (waves 0-1) ----------------
    if (tid < MROWS) {
        const int tl = tid >> 3;
        const int k  = tid & 7;
        float s = scoreS[tid];
        float mx = s;
        mx = fmaxf(mx, __shfl_xor(mx, 1, 32));
        mx = fmaxf(mx, __shfl_xor(mx, 2, 32));
        mx = fmaxf(mx, __shfl_xor(mx, 4, 32));
        const float e = __expf(s - mx);
        float sm = e;
        sm += __shfl_xor(sm, 1, 32);
        sm += __shfl_xor(sm, 2, 32);
        sm += __shfl_xor(sm, 4, 32);
        float w = e / sm;
        const int t   = t0 + tl;
        const int cti = ctp[b * T_ + t];
        if (cti < 0) w = 0.0f;
        lam[(b * T_ + t) * K_ + k] = w;
        wbuf[tid] = w;
        const int c = cti < 0 ? 0 : cti;
        aaS[tid] = aa[(b * K_ + k) * L_ + c];
    }
    __syncthreads();

    // ---------------- Phase 4: scatter weights into p_copy ----------------
    if (tid < TPB_T * V_) {
        const int tl = tid / V_;
        const int v  = tid - tl * V_;
        float s = 0.0f;
        #pragma unroll
        for (int k = 0; k < K_; ++k)
            s += (aaS[tl * 8 + k] == v) ? wbuf[tl * 8 + k] : 0.0f;
        p_copy[(b * T_ + t0 + tl) * V_ + v] = s;
    }
}

extern "C" void kernel_launch(void* const* d_in, const int* in_sizes, int n_in,
                              void* d_out, int out_size, void* d_ws, size_t ws_size,
                              hipStream_t stream) {
    const float* hs   = (const float*)d_in[0];
    const float* ex   = (const float*)d_in[1];
    const float* colf = (const float*)d_in[2];
    const int*   ctp  = (const int*)d_in[3];
    const int*   aa   = (const int*)d_in[4];
    const float* W1   = (const float*)d_in[5];
    const float* b1p  = (const float*)d_in[6];
    const float* W2   = (const float*)d_in[7];
    const float* b2p  = (const float*)d_in[8];

    _Float16* W1T = (_Float16*)d_ws;               // 256*448*2 = 229 KB scratch

    float* p_copy = (float*)d_out;                 // (B,T,V) first
    float* lam    = p_copy + (B_ * T_ * V_);       // then (B,T,K)

    // 1) transpose+convert W1 to f16 (idempotent, graph-safe)
    prep_w1t<<<(H_ * (FEAT / 2)) / 256, 256, 0, stream>>>(W1, W1T);

    // 2) fused gather + f16-WMMA MLP + softmax + scatter
    const int grid = B_ * (T_ / TPB_T);            // 1024 workgroups
    copyhead_fused<<<grid, 256, 0, stream>>>(hs, ex, colf, ctp, aa,
                                             W1T, b1p, W2, b2p, p_copy, lam);
}
